// GraphRes_738734375754
// MI455X (gfx1250) — matrix operand order repaired and checked by reading, not compiled, not run on hardware
//
#include <hip/hip_runtime.h>
#include <hip/hip_bf16.h>

typedef __attribute__((ext_vector_type(16))) _Float16 v16h;
typedef __attribute__((ext_vector_type(8)))  float    v8f;
typedef __attribute__((ext_vector_type(4)))  unsigned v4ui;
typedef __attribute__((ext_vector_type(8)))  int      v8i;
typedef __attribute__((ext_vector_type(4)))  int      v4i;

namespace gr {

constexpr int N      = 250000;
constexpr int E      = 1000000;
constexpr int B      = 25;
constexpr int NX     = 8, NY = 9, NC = NX * NY;   // 72 voxel cells / graph
constexpr int M      = B * NC;                    // 1800 pooled slots
constexpr long long MM = (long long)M * M;        // 3,240,000 adjacency slots
constexpr int GRIDS  = 16;
constexpr int NOUTC  = 2;
constexpr float EPS  = 1e-5f;

__device__ __forceinline__ void spline8(float p0, float p1, float p2, float* bas) {
  p0 = fminf(fmaxf(p0, 0.f), 1.f);
  p1 = fminf(fmaxf(p1, 0.f), 1.f);
  p2 = fminf(fmaxf(p2, 0.f), 1.f);
  const float q0 = 1.f - p0, q1 = 1.f - p1, q2 = 1.f - p2;
  bas[0] = q0*q1*q2; bas[1] = p0*q1*q2; bas[2] = q0*p1*q2; bas[3] = p0*p1*q2;
  bas[4] = q0*q1*p2; bas[5] = p0*q1*p2; bas[6] = q0*p1*p2; bas[7] = p0*p1*p2;
}

// TDM: async DMA of L contiguous f32 (global -> LDS), 1-D tensor descriptor.
// D# layout per CDNA5 ISA (groups 2/3 zero => <=2D tensor).
__device__ __forceinline__ void tdm_load_1d_f32(unsigned lds_off, const float* gptr, int L) {
  const unsigned long long ga = (unsigned long long)(uintptr_t)gptr;
  v4ui g0;
  g0[0] = 1u;                                         // count=1, load, no gather
  g0[1] = lds_off;                                    // lds_addr (bytes)
  g0[2] = (unsigned)(ga & 0xffffffffull);             // global_addr[31:0]
  g0[3] = (unsigned)((ga >> 32) & 0x01ffffffull)      // global_addr[56:32]
        | (2u << 30);                                 // type = 2 ("image")
  v8i g1;
  g1[0] = (int)(2u << 16);                            // wg_mask=0, data_size=4B
  g1[1] = (int)((unsigned)(L & 0xffff) << 16);        // tensor_dim0 lo16 @ [79:64]... [63:48]
  g1[2] = (int)(((unsigned)L >> 16) & 0xffffu)        // tensor_dim0 hi16
        | (1 << 16);                                  // tensor_dim1 = 1 (lo16)
  g1[3] = (int)((unsigned)(L & 0xffff) << 16);        // tile_dim0 = L  @ [127:112]
  g1[4] = 1;                                          // tile_dim1 = 1, tile_dim2 = 0
  g1[5] = L;                                          // tensor_dim0_stride lo32
  g1[6] = 0;                                          // stride hi16 / dim1_stride lo16
  g1[7] = 0;
  v4i z4; z4[0] = 0; z4[1] = 0; z4[2] = 0; z4[3] = 0;
#if defined(__clang_major__) && (__clang_major__ >= 23)
  v8i z8; for (int i = 0; i < 8; ++i) z8[i] = 0;
  __builtin_amdgcn_tensor_load_to_lds(g0, g1, z4, z4, z8, 0);
#else
  __builtin_amdgcn_tensor_load_to_lds(g0, g1, z4, z4, 0);
#endif
}

// ------------------------------------------------------------------ utilities
__global__ void zero32(unsigned* p, long long n) {
  long long i  = (long long)blockIdx.x * blockDim.x + threadIdx.x;
  long long st = (long long)gridDim.x * blockDim.x;
  for (; i < n; i += st) p[i] = 0u;
}

__global__ void init_maxv(unsigned* mv) {
  if (blockIdx.x == 0 && threadIdx.x == 0) *mv = __float_as_uint(1e-9f);
}

__global__ void degree_k(const int* __restrict__ dst, float* __restrict__ deg, int n) {
  int i = blockIdx.x * blockDim.x + threadIdx.x;
  if (i < n) atomicAdd(&deg[dst[i]], 1.f);
}

// ------------------------------------------------- WMMA edge-conv (conv1..5)
// z[e, s*CIN+c] = basis_s(e) * x[src(e), c];  msg = z @ Wflat  (Wflat = W as [8*CIN, COUT])
template<int CIN, int KPAD, int BASE>
__device__ __forceinline__ void stage_half(const float* bas, const float* xr, _Float16* row) {
  constexpr int K = 8 * CIN;
#pragma unroll
  for (int j = 0; j < KPAD / 2; ++j) {
    const int k = BASE + j;
    float v = 0.f;
    if (k < K) v = bas[k / CIN] * xr[k % CIN];
    row[k] = (_Float16)v;
  }
}

template<int CIN, int COUT, int KPAD>
__global__ void __launch_bounds__(128)
edge_conv_wmma(const float* __restrict__ xin, const float* __restrict__ eattr,
               const int* __restrict__ src, const int* __restrict__ dst,
               const float* __restrict__ W, float* __restrict__ msg)
{
  constexpr int KT  = KPAD / 32;          // WMMA K-steps
  constexpr int NCH = (COUT + 15) / 16;   // 16-col output chunks
  constexpr int L   = 8 * CIN * COUT;     // flattened weight element count
  __shared__ _Float16 Alds[4][16][KPAD];  // per-wave 16-edge A tile
  __shared__ _Float16 Wl[KT][NCH][32][16];// W in B-fragment layout (f16)
  __shared__ float    Wraw[L];            // TDM landing buffer (raw f32)

  const int tid  = threadIdx.x;
  const int lane = tid & 31;
  const int wave = tid >> 5;

  // --- async: TDM pulls the whole weight tensor into LDS while we build A ---
  if (wave == 0) {
    tdm_load_1d_f32((unsigned)(uintptr_t)&Wraw[0], W, L);
    __builtin_amdgcn_s_wait_tensorcnt(0);
  }

  // stage A tile: 2 lanes per edge, each covers half of the K range
  const int tile = blockIdx.x * 4 + wave;
  const int e0   = tile * 16;
  const int eloc = lane & 15;
  const int half = lane >> 4;
  const int e    = e0 + eloc;

  // prefetch next tile block's streamed edge attributes into cache
  if (e + 4096 < E) __builtin_prefetch(eattr + (long long)(e + 4096) * 3, 0, 1);

  float bas[8];
  spline8(eattr[e * 3 + 0], eattr[e * 3 + 1], eattr[e * 3 + 2], bas);
  float xr[CIN];
  {
    const float* xrow = xin + (long long)src[e] * CIN;
#pragma unroll
    for (int c = 0; c < CIN; ++c) xr[c] = xrow[c];
  }
  _Float16* row = &Alds[wave][eloc][0];
  if (half == 0) stage_half<CIN, KPAD, 0>(bas, xr, row);
  else           stage_half<CIN, KPAD, KPAD / 2>(bas, xr, row);

  __syncthreads();   // TDM data + A tiles visible to whole workgroup

  // convert raw f32 weights (LDS) into B-fragment layout (f16):
  // lane n&15 = column, lanes>=16 hold k +8, halves i<8 -> k i, i>=8 -> k 16+(i-8)
  constexpr int WTOT = KT * NCH * 32 * 16;
  for (int t = tid; t < WTOT; t += 128) {
    const int i    = t & 15;
    const int ln   = (t >> 4) & 31;
    const int rest = t >> 9;
    const int nc   = rest % NCH;
    const int kk   = rest / NCH;
    const int kg   = kk * 32 + ((ln >> 4) << 3) + i + ((i >> 3) << 3);
    const int n    = (ln & 15) + nc * 16;
    float v = 0.f;
    if (kg < 8 * CIN && n < COUT) v = Wraw[kg * COUT + n];
    Wl[kk][nc][ln][i] = (_Float16)v;
  }

  __syncthreads();

  // chained WMMA over K
  v8f acc[NCH] = {};
  const int m16 = lane & 15;
  const int ka  = (lane < 16) ? 0 : 8;   // A-frag K base per lane half
#pragma unroll
  for (int kk = 0; kk < KT; ++kk) {
    v16h a;
    const _Float16* ar = &Alds[wave][m16][kk * 32 + ka];
#pragma unroll
    for (int i = 0; i < 8; ++i) { a[i] = ar[i]; a[8 + i] = ar[16 + i]; }
#pragma unroll
    for (int nc = 0; nc < NCH; ++nc) {
      v16h b;
      const _Float16* br = &Wl[kk][nc][lane][0];
#pragma unroll
      for (int i = 0; i < 16; ++i) b[i] = br[i];
      acc[nc] = __builtin_amdgcn_wmma_f32_16x16x32_f16(
          false, a, false, b, (short)0, acc[nc], false, false);
    }
  }

  // C layout: VGPR r -> M = r + 8*(lane>=16), N = lane&15; scatter-add to dst
#pragma unroll
  for (int nc = 0; nc < NCH; ++nc) {
#pragma unroll
    for (int r = 0; r < 8; ++r) {
      const int mrow = r + ((lane >> 4) << 3);
      const int n    = (lane & 15) + nc * 16;
      if (n < COUT) {
        const int ed = e0 + mrow;
        atomicAdd(&msg[(long long)dst[ed] * COUT + n], acc[nc][r]);
      }
    }
  }
}

// --------------------------------------------------------- BatchNorm (full N)
__global__ void bn_reduce(const float* __restrict__ msg, const float* __restrict__ deg,
                          int n, int C, int log2C, float* __restrict__ accum) {
  extern __shared__ float sh[];  // [2*C]
  for (int t = threadIdx.x; t < 2 * C; t += blockDim.x) sh[t] = 0.f;
  __syncthreads();
  const int T   = gridDim.x * blockDim.x;
  const int tid = blockIdx.x * blockDim.x + threadIdx.x;
  const int c   = tid & (C - 1);
  float s1 = 0.f, s2 = 0.f;
  for (long long r = tid >> log2C; r < n; r += (T >> log2C)) {
    const float v = msg[r * C + c] / fmaxf(deg[r], 1.f);
    s1 += v; s2 += v * v;
  }
  atomicAdd(&sh[c], s1);
  atomicAdd(&sh[C + c], s2);
  __syncthreads();
  for (int t = threadIdx.x; t < 2 * C; t += blockDim.x) atomicAdd(&accum[t], sh[t]);
}

__global__ void bn_apply(const float* __restrict__ msg, const float* __restrict__ deg,
                         int n, int C, int log2C, const float* __restrict__ accum,
                         const float* __restrict__ gamma, const float* __restrict__ beta,
                         float* __restrict__ xout) {
  const long long i = (long long)blockIdx.x * blockDim.x + threadIdx.x;
  if (i >= (long long)n * C) return;
  const int c       = (int)(i & (C - 1));
  const long long r = i >> log2C;
  const float v    = msg[i] / fmaxf(deg[r], 1.f);
  const float mean = accum[c] / (float)n;
  const float var  = accum[C + c] / (float)n - mean * mean;
  const float y    = (v - mean) * rsqrtf(var + EPS) * gamma[c] + beta[c];
  xout[i] = fmaxf(y, 0.f);
}

// ------------------------------------------------------------------- pooling
__global__ void cluster_k(const float* __restrict__ pos, const int* __restrict__ batch,
                          int* __restrict__ cl, int n) {
  const int i = blockIdx.x * blockDim.x + threadIdx.x;
  if (i >= n) return;
  const int cx = min(max((int)floorf(pos[i * 3 + 0] / 16.f), 0), NX - 1);
  const int cy = min(max((int)floorf(pos[i * 3 + 1] / 12.f), 0), NY - 1);
  cl[i] = batch[i] * NC + cy * NX + cx;
}

__global__ void pool_accum(const float* __restrict__ x, const float* __restrict__ pos,
                           const int* __restrict__ cl, float* __restrict__ cnt,
                           unsigned* __restrict__ xpbits, float* __restrict__ ppsum, int n) {
  const int i = blockIdx.x * blockDim.x + threadIdx.x;
  if (i >= n) return;
  const int c = cl[i];
  atomicAdd(&cnt[c], 1.f);
#pragma unroll
  for (int d = 0; d < 3; ++d) atomicAdd(&ppsum[c * 3 + d], pos[i * 3 + d]);
  // x >= 0 (post-ReLU): uint-bit max == float max, zero init == empty->0
  for (int ch = 0; ch < 32; ++ch)
    atomicMax(&xpbits[c * 32 + ch], __float_as_uint(x[(long long)i * 32 + ch]));
}

__global__ void pool_div(const float* __restrict__ ppsum, const float* __restrict__ cnt,
                         float* __restrict__ pp) {
  const int i = blockIdx.x * blockDim.x + threadIdx.x;
  if (i >= M) return;
  const float inv = 1.f / fmaxf(cnt[i], 1.f);
#pragma unroll
  for (int d = 0; d < 3; ++d) pp[i * 3 + d] = ppsum[i * 3 + d] * inv;
}

// adjacency byte-map: coalesces duplicates + drops self loops for free
__global__ void mark_k(const int* __restrict__ src, const int* __restrict__ dst,
                       const int* __restrict__ cl, unsigned char* __restrict__ adj, int n) {
  const int i = blockIdx.x * blockDim.x + threadIdx.x;
  if (i >= n) return;
  const int a = cl[src[i]], b = cl[dst[i]];
  if (a != b) adj[(long long)a * M + b] = 1;
}

__global__ void maxd_k(const unsigned char* __restrict__ adj, const float* __restrict__ pp,
                       unsigned* __restrict__ mv) {
  long long i  = (long long)blockIdx.x * blockDim.x + threadIdx.x;
  long long st = (long long)gridDim.x * blockDim.x;
  for (; i < MM; i += st) {
    if (!adj[i]) continue;
    const int rs = (int)(i / M), cs = (int)(i % M);
#pragma unroll
    for (int d = 0; d < 3; ++d)
      atomicMax(mv, __float_as_uint(fabsf(pp[cs * 3 + d] - pp[rs * 3 + d])));
  }
}

__global__ void pdeg_k(const unsigned char* __restrict__ adj, float* __restrict__ degp) {
  long long i  = (long long)blockIdx.x * blockDim.x + threadIdx.x;
  long long st = (long long)gridDim.x * blockDim.x;
  for (; i < MM; i += st)
    if (adj[i]) atomicAdd(&degp[(int)(i % M)], 1.f);
}

// ------------------------------------------------ pooled SplineConv (conv6/7)
template<int CIN, int COUT>
__global__ void __launch_bounds__(256)
pooled_conv(const unsigned char* __restrict__ adj, const float* __restrict__ xin,
            const float* __restrict__ pp, const unsigned* __restrict__ mvbits,
            const float* __restrict__ W, float* __restrict__ msg) {
  __shared__ float Wl[8 * CIN * COUT];
  for (int t = threadIdx.x; t < 8 * CIN * COUT; t += blockDim.x) Wl[t] = W[t];
  __syncthreads();
  const float inv2 = 0.5f / __uint_as_float(*mvbits);

  long long i  = (long long)blockIdx.x * blockDim.x + threadIdx.x;
  long long st = (long long)gridDim.x * blockDim.x;
  for (; i < MM; i += st) {
    if (!adj[i]) continue;
    const int rs = (int)(i / M), cs = (int)(i % M);
    float bas[8];
    spline8((pp[cs * 3 + 0] - pp[rs * 3 + 0]) * inv2 + 0.5f,
            (pp[cs * 3 + 1] - pp[rs * 3 + 1]) * inv2 + 0.5f,
            (pp[cs * 3 + 2] - pp[rs * 3 + 2]) * inv2 + 0.5f, bas);
    float acc[COUT];
#pragma unroll
    for (int n = 0; n < COUT; ++n) acc[n] = 0.f;
    for (int c = 0; c < CIN; ++c) {
      const float xc = xin[rs * CIN + c];
#pragma unroll
      for (int s = 0; s < 8; ++s) {
        const float a = bas[s] * xc;
        const float* wr = &Wl[(s * CIN + c) * COUT];
#pragma unroll
        for (int n = 0; n < COUT; ++n) acc[n] += a * wr[n];
      }
    }
#pragma unroll
    for (int n = 0; n < COUT; ++n) atomicAdd(&msg[cs * COUT + n], acc[n]);
  }
}

// masked BatchNorm + ReLU on pooled graph (C=32, M=1800): one big block
__global__ void __launch_bounds__(1024)
bn_pool(const float* __restrict__ msg, const float* __restrict__ degp,
        const float* __restrict__ cntv, const float* __restrict__ gamma,
        const float* __restrict__ beta, float* __restrict__ xout) {
  __shared__ float ssum[32][32];
  __shared__ float ssq[32][32];
  __shared__ float smean[32], sinv[32], scnt;
  const int ch = threadIdx.x & 31, g = threadIdx.x >> 5;
  float s1 = 0.f, s2 = 0.f;
  for (int r = g; r < M; r += 32) {
    const float m = (cntv[r] > 0.f) ? 1.f : 0.f;
    const float v = msg[r * 32 + ch] / fmaxf(degp[r], 1.f);
    s1 += v * m; s2 += v * v * m;
  }
  ssum[g][ch] = s1; ssq[g][ch] = s2;
  __syncthreads();
  if (g == 0) {
    float a = 0.f, b = 0.f;
    for (int j = 0; j < 32; ++j) { a += ssum[j][ch]; b += ssq[j][ch]; }
    ssum[0][ch] = a; ssq[0][ch] = b;
    if (ch == 0) {
      float c = 0.f;
      for (int r = 0; r < M; ++r) c += (cntv[r] > 0.f) ? 1.f : 0.f;
      scnt = fmaxf(c, 1.f);
    }
  }
  __syncthreads();
  if (g == 0) {
    const float cnt  = scnt;
    const float mean = ssum[0][ch] / cnt;
    const float var  = ssq[0][ch] / cnt - mean * mean;
    smean[ch] = mean;
    sinv[ch]  = rsqrtf(var + EPS) * gamma[ch];
  }
  __syncthreads();
  for (int r = g; r < M; r += 32) {
    const float m = (cntv[r] > 0.f) ? 1.f : 0.f;
    const float v = msg[r * 32 + ch] / fmaxf(degp[r], 1.f);
    const float y = (v - smean[ch]) * sinv[ch] + beta[ch];
    xout[r * 32 + ch] = fmaxf(y, 0.f) * m;
  }
}

// ----------------------------------------------------- MaxPoolingX + FC head
__global__ void poolx_k(const float* __restrict__ xp, const float* __restrict__ pp,
                        const float* __restrict__ cntv, unsigned* __restrict__ xgbits) {
  const int i = blockIdx.x * blockDim.x + threadIdx.x;
  if (i >= M * 32) return;
  const int r = i >> 5, ch = i & 31;
  if (cntv[r] <= 0.f) return;
  const int bcl = r / NC;
  const int gx = min(max((int)floorf(pp[r * 3 + 0] / 30.f), 0), 3);
  const int gy = min(max((int)floorf(pp[r * 3 + 1] / 25.f), 0), 3);
  const int seg = bcl * GRIDS + gy * 4 + gx;
  atomicMax(&xgbits[seg * 32 + ch], __float_as_uint(xp[i]));  // xp >= 0
}

__global__ void fc_k(const float* __restrict__ xg, const float* __restrict__ w,
                     float* __restrict__ out) {
  const int i = threadIdx.x;
  if (i >= B * NOUTC) return;
  const int b = i / NOUTC, o = i % NOUTC;
  float s = 0.f;
  for (int j = 0; j < GRIDS * 32; ++j)
    s += xg[b * GRIDS * 32 + j] * w[o * GRIDS * 32 + j];
  out[i] = s;
}

}  // namespace gr

// =============================================================== entry point
extern "C" void kernel_launch(void* const* d_in, const int* in_sizes, int n_in,
                              void* d_out, int out_size, void* d_ws, size_t ws_size,
                              hipStream_t stream) {
  using namespace gr;
  const float* x0    = (const float*)d_in[0];
  const float* pos   = (const float*)d_in[1];
  const float* eattr = (const float*)d_in[2];
  const float* w[7];   for (int i = 0; i < 7; ++i) w[i] = (const float*)d_in[3 + i];
  const float* ga[7];  for (int i = 0; i < 7; ++i) ga[i] = (const float*)d_in[10 + i];
  const float* be[7];  for (int i = 0; i < 7; ++i) be[i] = (const float*)d_in[17 + i];
  const float* fc_w  = (const float*)d_in[24];
  const int*   eidx  = (const int*)d_in[25];
  const int*   batch = (const int*)d_in[26];
  const int* src = eidx;
  const int* dst = eidx + E;

  // workspace carve-up
  float* wsf   = (float*)d_ws;
  float* msg   = wsf;                         // N*32
  float* xA    = msg  + (size_t)N * 32;       // N*32
  float* xB    = xA   + (size_t)N * 32;       // N*32
  float* deg   = xB   + (size_t)N * 32;       // N
  float* accum = deg  + N;                    // 64
  float* cnt   = accum + 64;                  // M
  float* ppsum = cnt  + M;                    // 3M
  float* pp    = ppsum + 3 * M;               // 3M
  float* xp    = pp   + 3 * M;                // 32M
  float* xq    = xp   + 32 * M;               // 32M
  float* msgp  = xq   + 32 * M;               // 32M
  float* degp  = msgp + 32 * M;               // M
  float* xg    = degp + M;                    // B*GRIDS*32
  unsigned* maxv      = (unsigned*)(xg + B * GRIDS * 32);
  int*      cluster   = (int*)(maxv + 64);
  unsigned char* adj  = (unsigned char*)(cluster + N);  // M*M bytes (mult of 4)

  const int ZB = 2048, ZT = 256;

  // degree (same graph for all 5 full-graph convs)
  zero32<<<ZB, ZT, 0, stream>>>((unsigned*)deg, N);
  degree_k<<<(E + 255) / 256, 256, 0, stream>>>(dst, deg, E);

  struct Layer { int cin, cout, l2c; };
  const Layer L[5] = {{1,8,3},{8,16,4},{16,16,4},{16,16,4},{16,32,5}};
  const float* xin = x0;
  float* pingpong[2] = {xA, xB};
  for (int i = 0; i < 5; ++i) {
    float* xout = pingpong[i & 1];
    zero32<<<ZB, ZT, 0, stream>>>((unsigned*)msg, (long long)N * L[i].cout);
    switch (i) {
      case 0: edge_conv_wmma<1, 8, 32 ><<<E / 64, 128, 0, stream>>>(xin, eattr, src, dst, w[0], msg); break;
      case 1: edge_conv_wmma<8, 16, 64><<<E / 64, 128, 0, stream>>>(xin, eattr, src, dst, w[1], msg); break;
      case 2: edge_conv_wmma<16,16,128><<<E / 64, 128, 0, stream>>>(xin, eattr, src, dst, w[2], msg); break;
      case 3: edge_conv_wmma<16,16,128><<<E / 64, 128, 0, stream>>>(xin, eattr, src, dst, w[3], msg); break;
      case 4: edge_conv_wmma<16,32,128><<<E / 64, 128, 0, stream>>>(xin, eattr, src, dst, w[4], msg); break;
    }
    zero32<<<1, 64, 0, stream>>>((unsigned*)accum, 64);
    bn_reduce<<<1024, 256, 2 * L[i].cout * sizeof(float), stream>>>(msg, deg, N, L[i].cout, L[i].l2c, accum);
    bn_apply<<<((long long)N * L[i].cout + 255) / 256, 256, 0, stream>>>(
        msg, deg, N, L[i].cout, L[i].l2c, accum, ga[i], be[i], xout);
    xin = xout;
  }
  const float* x5 = xin;  // [N,32]

  // ---- MaxPooling to voxel grid ----
  cluster_k<<<(N + 255) / 256, 256, 0, stream>>>(pos, batch, cluster, N);
  zero32<<<ZB, ZT, 0, stream>>>((unsigned*)cnt, M);
  zero32<<<ZB, ZT, 0, stream>>>((unsigned*)ppsum, 3 * M);
  zero32<<<ZB, ZT, 0, stream>>>((unsigned*)xp, 32 * M);
  pool_accum<<<(N + 255) / 256, 256, 0, stream>>>(x5, pos, cluster, cnt, (unsigned*)xp, ppsum, N);
  pool_div<<<(M + 255) / 256, 256, 0, stream>>>(ppsum, cnt, pp);

  // ---- edge coarsening via adjacency byte-map ----
  zero32<<<ZB, ZT, 0, stream>>>((unsigned*)adj, MM / 4);
  mark_k<<<(E + 255) / 256, 256, 0, stream>>>(src, dst, cluster, adj, E);
  init_maxv<<<1, 32, 0, stream>>>(maxv);
  maxd_k<<<1024, 256, 0, stream>>>(adj, pp, maxv);
  zero32<<<ZB, ZT, 0, stream>>>((unsigned*)degp, M);
  pdeg_k<<<1024, 256, 0, stream>>>(adj, degp);

  // conv6
  zero32<<<ZB, ZT, 0, stream>>>((unsigned*)msgp, 32 * M);
  pooled_conv<32, 32><<<1024, 256, 0, stream>>>(adj, xp, pp, maxv, w[5], msgp);
  bn_pool<<<1, 1024, 0, stream>>>(msgp, degp, cnt, ga[5], be[5], xq);
  // conv7
  zero32<<<ZB, ZT, 0, stream>>>((unsigned*)msgp, 32 * M);
  pooled_conv<32, 32><<<1024, 256, 0, stream>>>(adj, xq, pp, maxv, w[6], msgp);
  bn_pool<<<1, 1024, 0, stream>>>(msgp, degp, cnt, ga[6], be[6], xp);

  // ---- MaxPoolingX + FC ----
  zero32<<<64, 256, 0, stream>>>((unsigned*)xg, B * GRIDS * 32);
  poolx_k<<<(M * 32 + 255) / 256, 256, 0, stream>>>(xp, pp, cnt, (unsigned*)xg);
  fc_k<<<1, 64, 0, stream>>>(xg, fc_w, (float*)d_out);
}
